// CompressCore_16655883174674
// MI455X (gfx1250) — matrix operand order repaired and checked by reading, not compiled, not run on hardware
//
#include <hip/hip_runtime.h>
#include <hip/hip_bf16.h>

typedef __attribute__((ext_vector_type(16))) __bf16 v16bf;
typedef __attribute__((ext_vector_type(8)))  __bf16 v8bf;
typedef __attribute__((ext_vector_type(8)))  float  v8f;

#define N_BATCH 4
#define C_IN    64
#define HH      256
#define WWID    256
#define C_OUT   32
#define HWSZ    65536
#define KSEL    3276
#define IDX_OFF  (N_BATCH*KSEL)              /* 13104 */
#define ENC_OFF  (IDX_OFF + N_BATCH*KSEL*2)  /* 39312 */

/* workspace layout (bytes) */
#define XB_BYTES   (N_BATCH*C_IN*HWSZ*2)         /* 33,554,432: bf16 NCHW input   */
#define WB_OFF_B   XB_BYTES
#define WB_BYTES   (9*32*64*2)                   /* 36,864: bf16 WMMA-layout wts  */
#define COMP_OFF_B (WB_OFF_B + WB_BYTES)         /* 33,591,296: f32 compressed    */

#define TCOLS 130                                /* halo columns per block: 128+2 */

__device__ __forceinline__ unsigned short f2bf(float f) {
    unsigned u = __float_as_uint(f);
    u += 0x7FFFu + ((u >> 16) & 1u);          // round-to-nearest-even
    return (unsigned short)(u >> 16);
}

__device__ __forceinline__ void wait_async0() {
#if __has_builtin(__builtin_amdgcn_s_wait_asynccnt)
    __builtin_amdgcn_s_wait_asynccnt(0);
#else
    asm volatile("s_wait_asynccnt 0x0" ::: "memory");
#endif
}

// async copy 16B: global (saddr + voff) -> LDS byte address ldsoff (ASYNCcnt-tracked)
__device__ __forceinline__ void async_g2lds_b128(unsigned ldsoff, unsigned voff, const void* saddr) {
    asm volatile("global_load_async_to_lds_b128 %0, %1, %2"
                 :: "v"(ldsoff), "v"(voff), "s"(saddr) : "memory");
}

// ---------------- Kernel 0a: X f32 NCHW -> bf16 (streaming, vectorized) ----------------
__global__ __launch_bounds__(256)
void cvt_input_kernel(const float* __restrict__ X, unsigned short* __restrict__ Xb) {
    const int stride = gridDim.x * blockDim.x;
    for (int i = blockIdx.x * blockDim.x + threadIdx.x; i < (N_BATCH*C_IN*HWSZ)/4; i += stride) {
        float4 v = reinterpret_cast<const float4*>(X)[i];
        unsigned long long p =  (unsigned long long)f2bf(v.x)
                             | ((unsigned long long)f2bf(v.y) << 16)
                             | ((unsigned long long)f2bf(v.z) << 32)
                             | ((unsigned long long)f2bf(v.w) << 48);
        reinterpret_cast<unsigned long long*>(Xb)[i] = p;
    }
}

// ---------------- Kernel 0b: weights -> bf16 in WMMA-B LDS layout [tap][o][c] ----------------
__global__ __launch_bounds__(256)
void cvt_weights_kernel(const float* __restrict__ Wt, unsigned short* __restrict__ wBg) {
    for (int idx = threadIdx.x; idx < 9*32*64; idx += 256) {
        int o = idx / 576; int r = idx % 576; int c = r / 9; int k = r % 9;
        wBg[(k*32 + o)*64 + c] = f2bf(Wt[idx]);
    }
}

// ---------------- Kernel 1: implicit-GEMM 3x3 conv via WMMA bf16, M-blocked x2 ----------------
// Block = 128 threads (4 waves): one row h, 128 consecutive w, all 32 output channels.
// Each wave: 32 positions x 32 channels = 2 M-tiles x 2 N-tiles; B fragments are
// reused across both M-tiles (72 v_wmma per wave, 8 LDS b128 reads per k-step).
// Weights arrive via global_load_async_to_lds_b128 (ASYNCcnt) while the VALU
// transposes the input halo tile [c][h][w] -> [row][col][c].
__global__ __launch_bounds__(128)
void conv_wmma_kernel(const unsigned short* __restrict__ Xb,
                      const unsigned short* __restrict__ wBg,
                      const float* __restrict__ Bias, float* __restrict__ enc,
                      float* __restrict__ comp) {
    __shared__ __align__(32) unsigned short inA[3*TCOLS*64];  // [row][col(-1..128)][c] bf16
    __shared__ __align__(32) unsigned short wB[9*32*64];      // [tap][o][c] bf16

    const int tid = threadIdx.x;
    const int pos_base = blockIdx.x * 128;
    const int n   = pos_base >> 16;
    const int rem = pos_base & 65535;
    const int h   = rem >> 8;
    const int w0  = rem & 255;          // 0 or 128

    // kick off async weight staging: 36,864 B = 2304 x b128, 18 per thread
    {
        unsigned ldsbase = (unsigned)(uintptr_t)(&wB[0]);   // low 32 bits = LDS address
        #pragma unroll
        for (int j = 0; j < 18; ++j) {
            unsigned boff = (unsigned)(tid + j*128) * 16u;
            async_g2lds_b128(ldsbase + boff, boff, (const void*)wBg);
        }
    }

    // stage input halo tile (bf16 source): iterate (c,row,col), col fastest => coalesced
    for (int idx = tid; idx < 64*3*TCOLS; idx += 128) {
        int c = idx / (3*TCOLS); int r2 = idx % (3*TCOLS); int row = r2 / TCOLS; int col = r2 % TCOLS;
        int hh = h + row - 1; int ww = w0 + col - 1;
        unsigned short v = 0;
        if (hh >= 0 && hh < HH && ww >= 0 && ww < WWID)
            v = Xb[((n*C_IN + c)*HH + hh)*WWID + ww];
        inA[(row*TCOLS + col)*64 + c] = v;
    }
    wait_async0();
    __syncthreads();

    const int wv   = tid >> 5;
    const int l    = tid & 31;
    const int m    = l & 15;
    const int msel = l >> 4;
    const int pm   = wv * 32;      // 32 positions per wave (2 M-tiles)
    const int baseA = msel * 16;   // A-frag K packing: contiguous 16
    const int baseB = msel * 16;   // B-frag K packing: contiguous 16

    v8f acc00 = {}, acc01 = {}, acc10 = {}, acc11 = {};

    #pragma unroll
    for (int it = 0; it < 18; ++it) {
        int s  = it >> 1;           // 3x3 tap
        int c0 = (it & 1) * 32;     // input-channel chunk
        int ky = s / 3, kx = s % 3;
        int colA = pm + m + kx;     // halo column for M-tile 0
        const unsigned short* pa0 = &inA[(ky*TCOLS + colA)*64      + c0 + baseA];
        const unsigned short* pa1 = &inA[(ky*TCOLS + colA + 16)*64 + c0 + baseA];
        v8bf a0lo = *reinterpret_cast<const v8bf*>(pa0);
        v8bf a0hi = *reinterpret_cast<const v8bf*>(pa0 + 8);
        v8bf a1lo = *reinterpret_cast<const v8bf*>(pa1);
        v8bf a1hi = *reinterpret_cast<const v8bf*>(pa1 + 8);
        v16bf av0 = __builtin_shufflevector(a0lo, a0hi, 0,1,2,3,4,5,6,7,8,9,10,11,12,13,14,15);
        v16bf av1 = __builtin_shufflevector(a1lo, a1hi, 0,1,2,3,4,5,6,7,8,9,10,11,12,13,14,15);
        v16bf bv0 = *reinterpret_cast<const v16bf*>(&wB[(s*32 + m)*64      + c0 + baseB]);
        v16bf bv1 = *reinterpret_cast<const v16bf*>(&wB[(s*32 + m + 16)*64 + c0 + baseB]);
        acc00 = __builtin_amdgcn_wmma_f32_16x16x32_bf16(false, av0, false, bv0, (short)0, acc00, false, false);
        acc01 = __builtin_amdgcn_wmma_f32_16x16x32_bf16(false, av0, false, bv1, (short)0, acc01, false, false);
        acc10 = __builtin_amdgcn_wmma_f32_16x16x32_bf16(false, av1, false, bv0, (short)0, acc10, false, false);
        acc11 = __builtin_amdgcn_wmma_f32_16x16x32_bf16(false, av1, false, bv1, (short)0, acc11, false, false);
    }

    const float b0 = Bias[m], b1 = Bias[m + 16];
    float a00[8], a01[8], a10[8], a11[8];
    #pragma unroll
    for (int r = 0; r < 8; ++r) {
        a00[r] = acc00[r] + b0; a01[r] = acc01[r] + b1;
        a10[r] = acc10[r] + b0; a11[r] = acc11[r] + b1;
    }

    // store encoded: VGPR r -> consecutive w -> vectorized b128 stores
    const int wbase0 = w0 + pm + msel*8;
    const int wbase1 = wbase0 + 16;
    {
        float* e0 = &enc[((n*C_OUT + m)*HH + h)*WWID];
        float* e1 = &enc[((n*C_OUT + m + 16)*HH + h)*WWID];
        float4 t;
        t = make_float4(a00[0],a00[1],a00[2],a00[3]); *(float4*)(e0 + wbase0)     = t;
        t = make_float4(a00[4],a00[5],a00[6],a00[7]); *(float4*)(e0 + wbase0 + 4) = t;
        t = make_float4(a10[0],a10[1],a10[2],a10[3]); *(float4*)(e0 + wbase1)     = t;
        t = make_float4(a10[4],a10[5],a10[6],a10[7]); *(float4*)(e0 + wbase1 + 4) = t;
        t = make_float4(a01[0],a01[1],a01[2],a01[3]); *(float4*)(e1 + wbase0)     = t;
        t = make_float4(a01[4],a01[5],a01[6],a01[7]); *(float4*)(e1 + wbase0 + 4) = t;
        t = make_float4(a11[0],a11[1],a11[2],a11[3]); *(float4*)(e1 + wbase1)     = t;
        t = make_float4(a11[4],a11[5],a11[6],a11[7]); *(float4*)(e1 + wbase1 + 4) = t;
    }

    // compressed = sum over 32 channels (channels live across the 16 lanes of each half)
    float* crow = &comp[n*HWSZ + h*WWID];
    #pragma unroll
    for (int r = 0; r < 8; ++r) {
        float s = a00[r] + a01[r];
        s += __shfl_xor(s, 1);
        s += __shfl_xor(s, 2);
        s += __shfl_xor(s, 4);
        s += __shfl_xor(s, 8);
        if (m == 0) crow[wbase0 + r] = s;
    }
    #pragma unroll
    for (int r = 0; r < 8; ++r) {
        float s = a10[r] + a11[r];
        s += __shfl_xor(s, 1);
        s += __shfl_xor(s, 2);
        s += __shfl_xor(s, 4);
        s += __shfl_xor(s, 8);
        if (m == 0) crow[wbase1 + r] = s;
    }
}

// ---------------- Kernel 2: per-batch top-3276 via 256KB-LDS histogram + bitonic sort ----------------
__device__ __forceinline__ unsigned monokey(float f) {
    unsigned u = __float_as_uint(f);
    return (u & 0x80000000u) ? ~u : (u | 0x80000000u);   // monotone: larger float -> larger key
}

__global__ __launch_bounds__(1024)
void topk_kernel(const float* __restrict__ comp, const int* __restrict__ perm,
                 float* __restrict__ feat, int* __restrict__ idxo) {
    __shared__ union {
        unsigned int hist[65536];            // 256 KB: top-16-bit key histogram
        unsigned long long cand[8192];       // reused: (key<<32)|~idx candidates
    } sh;
    __shared__ int sP;
    __shared__ unsigned sCnt;

    const int n = blockIdx.x;
    const int tid = threadIdx.x;
    const float* cf = comp + n*HWSZ;

    for (int i = tid; i < 65536; i += 1024) sh.hist[i] = 0u;
    __syncthreads();
    for (int i = tid; i < HWSZ; i += 1024)
        atomicAdd(&sh.hist[monokey(cf[i]) >> 16], 1u);
    __syncthreads();
    if (tid == 0) {                          // suffix scan from top to find rank-KSEL bin
        unsigned run = 0; int P = 0;
        for (int b = 65535; b >= 0; --b) {
            unsigned c = sh.hist[b];
            if (run + c >= KSEL) { P = b; break; }
            run += c;
        }
        sP = P; sCnt = 0;
    }
    __syncthreads();
    const unsigned P = (unsigned)sP;
    for (int i = tid; i < 8192; i += 1024) sh.cand[i] = 0ull;   // pad sorts last
    __syncthreads();
    for (int i = tid; i < HWSZ; i += 1024) {
        unsigned k = monokey(cf[i]);
        if ((k >> 16) >= P) {
            unsigned p = atomicAdd(&sCnt, 1u);
            if (p < 8192)
                sh.cand[p] = ((unsigned long long)k << 32) | (unsigned)(~i); // tie: smaller idx wins
        }
    }
    __syncthreads();
    // bitonic sort, descending composite -> deterministic order
    for (unsigned kk = 2; kk <= 8192; kk <<= 1) {
        for (unsigned j = kk >> 1; j > 0; j >>= 1) {
            for (unsigned idx = tid; idx < 8192; idx += 1024) {
                unsigned ixj = idx ^ j;
                if (ixj > idx) {
                    bool desc = ((idx & kk) == 0);
                    unsigned long long x = sh.cand[idx], y = sh.cand[ixj];
                    if (desc ? (x < y) : (x > y)) { sh.cand[idx] = y; sh.cand[ixj] = x; }
                }
            }
            __syncthreads();
        }
    }
    // output row after batch permutation: row p holds batch perm[p]
    int p = 0;
    #pragma unroll
    for (int q = 0; q < 4; ++q) if (perm[q] == n) p = q;
    for (int j = tid; j < KSEL; j += 1024) {
        unsigned long long c = sh.cand[j];
        unsigned i = ~(unsigned)c;
        feat[p*KSEL + j] = cf[i];
        idxo[(p*KSEL + j)*2 + 0] = (int)(i >> 8);   // h
        idxo[(p*KSEL + j)*2 + 1] = (int)(i & 255);  // w
    }
}

extern "C" void kernel_launch(void* const* d_in, const int* in_sizes, int n_in,
                              void* d_out, int out_size, void* d_ws, size_t ws_size,
                              hipStream_t stream) {
    (void)in_sizes; (void)n_in; (void)out_size; (void)ws_size;
    const float* X    = (const float*)d_in[0];
    const float* Wt   = (const float*)d_in[1];
    const float* Bias = (const float*)d_in[2];
    const int*   perm = (const int*)d_in[3];

    float* out  = (float*)d_out;
    float* feat = out;                       // [4][3276] f32
    int*   idxo = (int*)(out + IDX_OFF);     // [4][3276][2] i32 (bitwise)
    float* enc  = out + ENC_OFF;             // [4][32][256][256] f32

    char* ws = (char*)d_ws;
    unsigned short* Xb  = (unsigned short*)ws;                // bf16 NCHW input
    unsigned short* wBg = (unsigned short*)(ws + WB_OFF_B);   // bf16 WMMA-layout weights
    float*          comp = (float*)(ws + COMP_OFF_B);         // f32 compressed

    cvt_input_kernel  <<<dim3(2048), dim3(256), 0, stream>>>(X, Xb);
    cvt_weights_kernel<<<dim3(1),    dim3(256), 0, stream>>>(Wt, wBg);
    conv_wmma_kernel  <<<dim3(2048), dim3(128), 0, stream>>>(Xb, wBg, Bias, enc, comp);
    topk_kernel       <<<dim3(4),    dim3(1024), 0, stream>>>(comp, perm, feat, idxo);
}
